// ViT_44401371906285
// MI455X (gfx1250) — matrix operand order repaired and checked by reading, not compiled, not run on hardware
//
#include <hip/hip_runtime.h>

// ----- problem dimensions -----
#define N_REAL 3528            // b*n = 8*441
#define NPAD   3584            // N padded to multiple of 256 (WMMA tiling)
#define EMB    384             // embed_dim
#define DQK    512             // inner (q/k width)
#define DIM    389             // embed_dim + CLS_DIM
#define DPAD   448             // DIM padded to multiple of 64
#define MASKW  (NPAD / 32)     // 112 mask words per row
#define TOPK_K 64
#define NS_ITERS 25            // Newton-Schulz iterations
#define EPSF   2.2204460492503131e-16f

#define LDS_STRIDE 40          // bf16 elems per staged B row (80 bytes)
#define LDS_BUF    (64 * LDS_STRIDE)

typedef __bf16 bf16_t;
typedef __attribute__((ext_vector_type(16))) bf16_t v16bf;
typedef __attribute__((ext_vector_type(8)))  bf16_t v8bf;
typedef __attribute__((ext_vector_type(8)))  float  v8f;
typedef __attribute__((ext_vector_type(4)))  int    v4i;

#if defined(__has_builtin)
#if __has_builtin(__builtin_amdgcn_global_load_async_to_lds_b128) && \
    __has_builtin(__builtin_amdgcn_s_wait_asynccnt)
#define USE_ASYNC_LDS 1
#endif
#endif
#ifndef USE_ASYNC_LDS
#define USE_ASYNC_LDS 0
#endif

union FragU { v16bf v; v8bf h[2]; };

// A-fragment (16x32 bf16). lane r = lane&15 holds row (m0+r).
// lanes 0-15 hold K = {0..7, 16..23}, lanes 16-31 K = {8..15, 24..31}.
__device__ __forceinline__ v16bf load_a_frag(const bf16_t* A, int row, int ld,
                                             int k0, int half) {
  const bf16_t* p = A + (size_t)row * ld + k0 + half * 8;
  FragU u;
  u.h[0] = *(const v8bf*)(p);
  u.h[1] = *(const v8bf*)(p + 16);
  return u.v;
}
// B-fragment (32x16 bf16) from the LDS-staged tile (row = output column).
// lanes 0-15 hold K = {0..7, 16..23}, lanes 16-31 K = {8..15, 24..31}.
__device__ __forceinline__ v16bf load_b_frag_lds(const bf16_t* buf, int colLocal,
                                                 int half) {
  const bf16_t* p = buf + colLocal * LDS_STRIDE + half * 8;
  FragU u;
  u.h[0] = *(const v8bf*)(p);
  u.h[1] = *(const v8bf*)(p + 16);
  return u.v;
}
__device__ __forceinline__ v8f wmma_bf16(v16bf a, v16bf b, v8f c) {
  return __builtin_amdgcn_wmma_f32_16x16x32_bf16(false, a, false, b,
                                                 (short)0, c, false, false);
}
__device__ __forceinline__ v8f vzero8() {
  v8f z = {0.f, 0.f, 0.f, 0.f, 0.f, 0.f, 0.f, 0.f};
  return z;
}

// stage 16 bytes of the shared B tile: global -> LDS
__device__ __forceinline__ void stage_b16(const bf16_t* src, bf16_t* dst) {
#if USE_ASYNC_LDS
  // generic LDS pointer: low 32 bits are the LDS byte offset (aperture rule)
  __builtin_amdgcn_global_load_async_to_lds_b128(
      (__attribute__((address_space(1))) v4i*)(uintptr_t)src,
      (__attribute__((address_space(3))) v4i*)(uint32_t)(uintptr_t)dst, 0, 0);
#else
  *(v8bf*)dst = *(const v8bf*)src;
#endif
}
__device__ __forceinline__ void stage_wait(bool keep_one) {
#if USE_ASYNC_LDS
  if (keep_one) __builtin_amdgcn_s_wait_asynccnt(1);
  else          __builtin_amdgcn_s_wait_asynccnt(0);
#else
  (void)keep_one;
#endif
  __syncthreads();
}

// Core GEMM tile: block = 256 threads = 8 waves, each wave computes 32x64,
// block computes 256x64.  B tile (64 cols x 32 K) double-buffered in LDS.
// acc[0..3] = rows m0w..m0w+15, acc[4..7] = rows m0w+16..m0w+31.
__device__ __forceinline__ void gemm_tile(const bf16_t* __restrict__ A, int lda,
                                          const bf16_t* __restrict__ Bt, int ldb,
                                          int K, int m0w, int n0,
                                          bf16_t* sB, v8f acc[8]) {
  const int tid = threadIdx.x;
  const int lane = tid & 31, r = lane & 15, half = lane >> 4;
  const int srow = tid >> 2;      // 0..63 : staged column (row of Bt)
  const int sseg = tid & 3;       // 0..3  : 16-byte segment within 64B of K
#pragma unroll
  for (int t = 0; t < 8; ++t) acc[t] = vzero8();

  const int KS = K >> 5;
  // prologue: stage first tile into buffer 0
  stage_b16(Bt + (size_t)(n0 + srow) * ldb + sseg * 8,
            sB + srow * LDS_STRIDE + sseg * 8);
  for (int ks = 0; ks < KS; ++ks) {
    const int k0 = ks << 5;
    bf16_t* cur = sB + (ks & 1) * LDS_BUF;
    const bool more = (ks + 1) < KS;
    if (more)
      stage_b16(Bt + (size_t)(n0 + srow) * ldb + (k0 + 32) + sseg * 8,
                sB + ((ks + 1) & 1) * LDS_BUF + srow * LDS_STRIDE + sseg * 8);
    stage_wait(more);  // cur tile resident for all waves

    v16bf a0 = load_a_frag(A, m0w + r, lda, k0, half);
    v16bf a1 = load_a_frag(A, m0w + 16 + r, lda, k0, half);
    if (more) {
      __builtin_prefetch(A + (size_t)(m0w + r) * lda + k0 + 32, 0, 1);
      __builtin_prefetch(A + (size_t)(m0w + 16 + r) * lda + k0 + 32, 0, 1);
    }
#pragma unroll
    for (int t = 0; t < 4; ++t) {
      v16bf b = load_b_frag_lds(cur, t * 16 + r, half);
      acc[t]     = wmma_bf16(a0, b, acc[t]);
      acc[4 + t] = wmma_bf16(a1, b, acc[4 + t]);
    }
    __syncthreads();  // cur may be overwritten next iteration
  }
}

// ---------------- prep kernels ----------------

__global__ void prep_feat(const float* __restrict__ x, bf16_t* __restrict__ feat) {
  int i = blockIdx.x;
  for (int c = threadIdx.x; c < EMB; c += 128)
    feat[(size_t)i * EMB + c] =
        (i < N_REAL) ? (bf16_t)x[(size_t)i * DIM + c] : (bf16_t)0.f;
}

__global__ void prep_w(const float* __restrict__ Wq, const float* __restrict__ Wk,
                       bf16_t* __restrict__ wqT, bf16_t* __restrict__ wkT) {
  const float* W = blockIdx.z ? Wk : Wq;
  bf16_t* T = blockIdx.z ? wkT : wqT;
  int o = blockIdx.x;
  for (int c = threadIdx.x; c < EMB; c += 128)
    T[(size_t)o * EMB + c] = (bf16_t)W[(size_t)c * DQK + o];
}

__global__ void prep_xfT(const float* __restrict__ x, bf16_t* __restrict__ xfT) {
  int c = blockIdx.x;
  for (int i = threadIdx.x; i < NPAD; i += 256)
    xfT[(size_t)c * NPAD + i] =
        (i < N_REAL && c < DIM) ? (bf16_t)x[(size_t)i * DIM + c] : (bf16_t)0.f;
}

// ---------------- GEMM: q = feat@Wq, k = feat@Wk ----------------
__global__ void __launch_bounds__(256)
gemm_qk(const bf16_t* __restrict__ feat, const bf16_t* __restrict__ wqT,
        const bf16_t* __restrict__ wkT, bf16_t* __restrict__ qout,
        bf16_t* __restrict__ kout) {
  __shared__ bf16_t sB[2 * LDS_BUF];
  const int wave = threadIdx.x >> 5, lane = threadIdx.x & 31;
  const int r = lane & 15, half = lane >> 4;
  const bf16_t* Bt = blockIdx.z ? wkT : wqT;
  bf16_t* out = blockIdx.z ? kout : qout;
  const int m0w = blockIdx.x * 256 + wave * 32;
  const int n0 = blockIdx.y * 64;
  v8f acc[8];
  gemm_tile(feat, EMB, Bt, EMB, EMB, m0w, n0, sB, acc);
#pragma unroll
  for (int s = 0; s < 2; ++s)
#pragma unroll
    for (int t = 0; t < 4; ++t) {
      int n = n0 + t * 16 + r;
#pragma unroll
      for (int g = 0; g < 8; ++g) {
        int m = m0w + s * 16 + half * 8 + g;
        out[(size_t)m * DQK + n] = (bf16_t)acc[s * 4 + t][g];
      }
    }
}

__global__ void __launch_bounds__(128)
row_sumsq(const bf16_t* __restrict__ q, const bf16_t* __restrict__ k,
          float* __restrict__ sq, float* __restrict__ sk) {
  __shared__ float red[128];
  const bf16_t* src = blockIdx.z ? k : q;
  float* dst = blockIdx.z ? sk : sq;
  int row = blockIdx.x;
  float s = 0.f;
  for (int c = threadIdx.x; c < DQK; c += 128) {
    float v = (float)src[(size_t)row * DQK + c];
    s += v * v;
  }
  red[threadIdx.x] = s; __syncthreads();
  for (int st = 64; st > 0; st >>= 1) {
    if (threadIdx.x < st) red[threadIdx.x] += red[threadIdx.x + st];
    __syncthreads();
  }
  if (threadIdx.x == 0) dst[row] = red[0];
}

// ---------------- GEMM: dots = (sq + sk - 2 q k^T)/DQK ----------------
__global__ void __launch_bounds__(256)
gemm_dots(const bf16_t* __restrict__ q, const bf16_t* __restrict__ kt,
          const float* __restrict__ sq, const float* __restrict__ sk,
          float* __restrict__ dots) {
  __shared__ bf16_t sB[2 * LDS_BUF];
  const int wave = threadIdx.x >> 5, lane = threadIdx.x & 31;
  const int r = lane & 15, half = lane >> 4;
  const int m0w = blockIdx.x * 256 + wave * 32;
  const int n0 = blockIdx.y * 64;
  v8f acc[8];
  gemm_tile(q, DQK, kt, DQK, DQK, m0w, n0, sB, acc);
  const float invD = 1.f / (float)DQK;
#pragma unroll
  for (int s = 0; s < 2; ++s)
#pragma unroll
    for (int t = 0; t < 4; ++t) {
      int n = n0 + t * 16 + r;
#pragma unroll
      for (int g = 0; g < 8; ++g) {
        int m = m0w + s * 16 + half * 8 + g;
        dots[(size_t)m * NPAD + n] =
            (sq[m] + sk[n] - 2.f * acc[s * 4 + t][g]) * invD;
      }
    }
}

// ---------------- top-64 per row -> symmetric bitmask ----------------
__global__ void __launch_bounds__(256)
topk_mask(const float* __restrict__ dots, unsigned* __restrict__ maskw) {
  __shared__ float vals[N_REAL];
  __shared__ float smax[256];
  __shared__ int   sidx[256];
  const int i = blockIdx.x;
  for (int j = threadIdx.x; j < N_REAL; j += 256)
    vals[j] = dots[(size_t)i * NPAD + j];
  __syncthreads();
  for (int t = 0; t < TOPK_K; ++t) {
    float best = -3.4e38f; int bi = 0;
    for (int j = threadIdx.x; j < N_REAL; j += 256) {
      float v = vals[j];
      if (v > best) { best = v; bi = j; }
    }
    smax[threadIdx.x] = best; sidx[threadIdx.x] = bi;
    __syncthreads();
    for (int st = 128; st > 0; st >>= 1) {
      if (threadIdx.x < st && smax[threadIdx.x + st] > smax[threadIdx.x]) {
        smax[threadIdx.x] = smax[threadIdx.x + st];
        sidx[threadIdx.x] = sidx[threadIdx.x + st];
      }
      __syncthreads();
    }
    if (threadIdx.x == 0) {
      int w = sidx[0];
      vals[w] = -3.4e38f;
      atomicOr(&maskw[(size_t)i * MASKW + (w >> 5)], 1u << (w & 31));
      atomicOr(&maskw[(size_t)w * MASKW + (i >> 5)], 1u << (i & 31));
    }
    __syncthreads();
  }
}

// ---------------- B = I - alpha*(dots*mask) + EPS (bf16, zero-padded) -----
__global__ void __launch_bounds__(256)
build_B(const float* __restrict__ dots, const unsigned* __restrict__ maskw,
        const float* __restrict__ alpha_p, bf16_t* __restrict__ Bm) {
  size_t idx = (size_t)blockIdx.x * 256 + threadIdx.x;
  int i = (int)(idx / NPAD), j = (int)(idx % NPAD);
  float v = 0.f;
  if (i < N_REAL && j < N_REAL) {
    unsigned bit = (maskw[(size_t)i * MASKW + (j >> 5)] >> (j & 31)) & 1u;
    float a = bit ? dots[idx] : 0.f;
    v = ((i == j) ? 1.f : 0.f) - alpha_p[0] * a + EPSF;
  }
  Bm[idx] = (bf16_t)v;
}

__global__ void __launch_bounds__(128)
abs_rowcol(const bf16_t* __restrict__ Bm, float* __restrict__ rsum,
           float* __restrict__ csum) {
  __shared__ float red[128];
  int i = blockIdx.x;
  float s = 0.f;
  if (blockIdx.z == 0) {
    for (int j = threadIdx.x; j < NPAD; j += 128)
      s += fabsf((float)Bm[(size_t)i * NPAD + j]);
  } else {
    for (int j = threadIdx.x; j < NPAD; j += 128)
      s += fabsf((float)Bm[(size_t)j * NPAD + i]);
  }
  red[threadIdx.x] = s; __syncthreads();
  for (int st = 64; st > 0; st >>= 1) {
    if (threadIdx.x < st) red[threadIdx.x] += red[threadIdx.x + st];
    __syncthreads();
  }
  if (threadIdx.x == 0) (blockIdx.z ? csum : rsum)[i] = red[0];
}

__global__ void __launch_bounds__(256)
compute_scale(const float* __restrict__ rsum, const float* __restrict__ csum,
              float* __restrict__ scale) {
  __shared__ float r1[256], r2[256];
  float a = 0.f, b = 0.f;
  for (int i = threadIdx.x; i < NPAD; i += 256) {
    a = fmaxf(a, rsum[i]);
    b = fmaxf(b, csum[i]);
  }
  r1[threadIdx.x] = a; r2[threadIdx.x] = b; __syncthreads();
  for (int st = 128; st > 0; st >>= 1) {
    if (threadIdx.x < st) {
      r1[threadIdx.x] = fmaxf(r1[threadIdx.x], r1[threadIdx.x + st]);
      r2[threadIdx.x] = fmaxf(r2[threadIdx.x], r2[threadIdx.x + st]);
    }
    __syncthreads();
  }
  if (threadIdx.x == 0) scale[0] = 1.f / (r1[0] * r2[0]);
}

// X0 = B^T * s  (and its transpose)
__global__ void __launch_bounds__(256)
init_X(const bf16_t* __restrict__ Bm, const float* __restrict__ scale,
       bf16_t* __restrict__ X, bf16_t* __restrict__ XT) {
  size_t idx = (size_t)blockIdx.x * 256 + threadIdx.x;
  int i = (int)(idx / NPAD), j = (int)(idx % NPAD);
  float s = scale[0];
  X[idx]  = (bf16_t)(s * (float)Bm[(size_t)j * NPAD + i]);
  XT[idx] = (bf16_t)(s * (float)Bm[idx]);
}

// ---------------- Newton-Schulz GEMM 1: T^T = (B X)^T ----------------
__global__ void __launch_bounds__(256)
gemm_ns1(const bf16_t* __restrict__ Bm, const bf16_t* __restrict__ XT,
         bf16_t* __restrict__ Tt) {
  __shared__ bf16_t sB[2 * LDS_BUF];
  const int wave = threadIdx.x >> 5, lane = threadIdx.x & 31;
  const int r = lane & 15, half = lane >> 4;
  const int m0w = blockIdx.x * 256 + wave * 32;
  const int n0 = blockIdx.y * 64;
  v8f acc[8];
  gemm_tile(Bm, NPAD, XT, NPAD, NPAD, m0w, n0, sB, acc);
#pragma unroll
  for (int s = 0; s < 2; ++s)
#pragma unroll
    for (int t = 0; t < 4; ++t) {
      int n = n0 + t * 16 + r;
#pragma unroll
      for (int g = 0; g < 8; ++g) {
        int m = m0w + s * 16 + half * 8 + g;
        Tt[(size_t)n * NPAD + m] = (bf16_t)acc[s * 4 + t][g];
      }
    }
}

// ---------------- Newton-Schulz GEMM 2: X' = 2X - X T ----------------
__global__ void __launch_bounds__(256)
gemm_ns2(const bf16_t* __restrict__ X, const bf16_t* __restrict__ Tt,
         bf16_t* __restrict__ Xn, bf16_t* __restrict__ XnT) {
  __shared__ bf16_t sB[2 * LDS_BUF];
  const int wave = threadIdx.x >> 5, lane = threadIdx.x & 31;
  const int r = lane & 15, half = lane >> 4;
  const int m0w = blockIdx.x * 256 + wave * 32;
  const int n0 = blockIdx.y * 64;
  v8f acc[8];
  gemm_tile(X, NPAD, Tt, NPAD, NPAD, m0w, n0, sB, acc);
#pragma unroll
  for (int s = 0; s < 2; ++s)
#pragma unroll
    for (int t = 0; t < 4; ++t) {
      int n = n0 + t * 16 + r;
#pragma unroll
      for (int g = 0; g < 8; ++g) {
        int m = m0w + s * 16 + half * 8 + g;
        float v = 2.f * (float)X[(size_t)m * NPAD + n] - acc[s * 4 + t][g];
        Xn[(size_t)m * NPAD + n]  = (bf16_t)v;
        XnT[(size_t)n * NPAD + m] = (bf16_t)v;
      }
    }
}

// ---------------- out = X @ xf ----------------
__global__ void __launch_bounds__(256)
gemm_out(const bf16_t* __restrict__ X, const bf16_t* __restrict__ xfT,
         float* __restrict__ y) {
  __shared__ bf16_t sB[2 * LDS_BUF];
  const int wave = threadIdx.x >> 5, lane = threadIdx.x & 31;
  const int r = lane & 15, half = lane >> 4;
  const int m0w = blockIdx.x * 256 + wave * 32;
  const int n0 = blockIdx.y * 64;
  v8f acc[8];
  gemm_tile(X, NPAD, xfT, NPAD, NPAD, m0w, n0, sB, acc);
#pragma unroll
  for (int s = 0; s < 2; ++s)
#pragma unroll
    for (int t = 0; t < 4; ++t) {
      int n = n0 + t * 16 + r;
#pragma unroll
      for (int g = 0; g < 8; ++g) {
        int m = m0w + s * 16 + half * 8 + g;
        y[(size_t)m * DPAD + n] = acc[s * 4 + t][g];
      }
    }
}

__global__ void final_copy(const float* __restrict__ y, float* __restrict__ out) {
  int i = blockIdx.x;
  for (int c = threadIdx.x; c < DIM; c += 128)
    out[(size_t)i * DIM + c] = y[(size_t)i * DPAD + c];
}

// ---------------- host launcher ----------------
extern "C" void kernel_launch(void* const* d_in, const int* in_sizes, int n_in,
                              void* d_out, int out_size, void* d_ws, size_t ws_size,
                              hipStream_t stream) {
  const float* x     = (const float*)d_in[0];
  const float* Wq    = (const float*)d_in[1];
  const float* Wk    = (const float*)d_in[2];
  const float* alpha = (const float*)d_in[3];
  float* out = (float*)d_out;

  char* base = (char*)d_ws;
  size_t off = 0;
  auto take = [&](size_t bytes) -> char* {
    char* p = base + off;
    off += (bytes + 255) & ~((size_t)255);
    return p;
  };
  const size_t NN2 = (size_t)NPAD * NPAD * sizeof(bf16_t);

  bf16_t*   feat  = (bf16_t*)take((size_t)NPAD * EMB * 2);
  bf16_t*   wqT   = (bf16_t*)take((size_t)DQK * EMB * 2);
  bf16_t*   wkT   = (bf16_t*)take((size_t)DQK * EMB * 2);
  bf16_t*   qb    = (bf16_t*)take((size_t)NPAD * DQK * 2);
  bf16_t*   kb    = (bf16_t*)take((size_t)NPAD * DQK * 2);
  float*    sq    = (float*)take((size_t)NPAD * 4);
  float*    sk    = (float*)take((size_t)NPAD * 4);
  float*    dots  = (float*)take((size_t)NPAD * NPAD * 4);
  unsigned* maskw = (unsigned*)take((size_t)NPAD * MASKW * 4);
  bf16_t*   Bm    = (bf16_t*)take(NN2);
  float*    rsum  = (float*)take((size_t)NPAD * 4);
  float*    csum  = (float*)take((size_t)NPAD * 4);
  float*    scale = (float*)take(256);
  bf16_t*   Xa    = (bf16_t*)take(NN2);
  bf16_t*   XaT   = (bf16_t*)take(NN2);
  bf16_t*   Xb    = (bf16_t*)take(NN2);
  bf16_t*   XbT   = (bf16_t*)take(NN2);
  bf16_t*   Tt    = (bf16_t*)take(NN2);
  bf16_t*   xfT   = (bf16_t*)take((size_t)DPAD * NPAD * 2);
  float*    yf    = (float*)take((size_t)NPAD * DPAD * 4);

  (void)hipMemsetAsync(maskw, 0, (size_t)NPAD * MASKW * 4, stream);

  prep_feat<<<NPAD, 128, 0, stream>>>(x, feat);
  prep_w<<<dim3(DQK, 1, 2), 128, 0, stream>>>(Wq, Wk, wqT, wkT);
  prep_xfT<<<DPAD, 256, 0, stream>>>(x, xfT);

  gemm_qk<<<dim3(NPAD / 256, DQK / 64, 2), 256, 0, stream>>>(feat, wqT, wkT,
                                                             qb, kb);
  row_sumsq<<<dim3(NPAD, 1, 2), 128, 0, stream>>>(qb, kb, sq, sk);
  gemm_dots<<<dim3(NPAD / 256, NPAD / 64), 256, 0, stream>>>(qb, kb, sq, sk,
                                                             dots);

  topk_mask<<<N_REAL, 256, 0, stream>>>(dots, maskw);
  build_B<<<(unsigned)(((size_t)NPAD * NPAD) / 256), 256, 0, stream>>>(
      dots, maskw, alpha, Bm);
  abs_rowcol<<<dim3(NPAD, 1, 2), 128, 0, stream>>>(Bm, rsum, csum);
  compute_scale<<<1, 256, 0, stream>>>(rsum, csum, scale);
  init_X<<<(unsigned)(((size_t)NPAD * NPAD) / 256), 256, 0, stream>>>(
      Bm, scale, Xa, XaT);

  bf16_t *Xc = Xa, *XcT = XaT, *Xn = Xb, *XnT = XbT;
  for (int it = 0; it < NS_ITERS; ++it) {
    gemm_ns1<<<dim3(NPAD / 256, NPAD / 64), 256, 0, stream>>>(Bm, XcT, Tt);
    gemm_ns2<<<dim3(NPAD / 256, NPAD / 64), 256, 0, stream>>>(Xc, Tt, Xn, XnT);
    bf16_t* t0 = Xc; Xc = Xn; Xn = t0;
    bf16_t* t1 = XcT; XcT = XnT; XnT = t1;
  }

  gemm_out<<<dim3(NPAD / 256, DPAD / 64), 256, 0, stream>>>(Xc, xfT, yf);
  final_copy<<<N_REAL, 128, 0, stream>>>(yf, out);
}
